// DPFPAttention_12223476924538
// MI455X (gfx1250) — matrix operand order repaired and checked by reading, not compiled, not run on hardware
//
#include <hip/hip_runtime.h>
#include <hip/hip_bf16.h>

// ---------------------------------------------------------------------------
// DPFP linear attention (global delta-rule update) for MI455X / gfx1250.
// All GEMMs run through v_wmma_f32_16x16x32_f16 (wave32, f32 accumulate).
// Intermediates are f16 to halve HBM traffic (problem is bandwidth bound:
// ~47 GFLOP vs ~0.5 GB traffic @ 23.3 TB/s). Small weight matrices are
// pre-transposed to [N][K] so B-fragments are two contiguous 8-f16 reads that
// stay resident in the 192 MB L2.
// ---------------------------------------------------------------------------

typedef _Float16 f16;
typedef __attribute__((ext_vector_type(16))) _Float16 v16h;
typedef __attribute__((ext_vector_type(8)))  _Float16 v8h;
typedef __attribute__((ext_vector_type(8)))  float    v8f;

union HFrag { v16h v; v8h h[2]; };

#define B_   4
#define N_   4096
#define DIM_ 512
#define H_   8
#define D_   64
#define S_   128           // dpfp feature dim = 2*NU*D = 128
#define M_   (B_ * N_)     // 16384
#define QKV_ 1536

// ---------------------------------------------------------------------------
// Generic batched WMMA GEMM:  C = scale * (A @ BT^T) [+ addC] [+ bias]
//   A : [M][K] f16 row-major        (lda, batch stride sA)
//   BT: [N][K] f16 row-major        (ldb, batch stride sB)   i.e. B transposed
//   C : f32 (Cf) or f16 (Ch), ldc; batch offset = (z/cmod)*sCouter + (z%cmod)*sCinner
// Each wave computes a 16x64 strip (4 WMMA tiles); block = 8 waves = 128 M rows.
// ---------------------------------------------------------------------------
__global__ void __launch_bounds__(256)
wmma_gemm_bt(const f16* __restrict__ A, const f16* __restrict__ BT,
             float* __restrict__ Cf, f16* __restrict__ Ch,
             const float* __restrict__ addC, const float* __restrict__ bias,
             int M, int N, int K, int lda, int ldb, int ldc,
             long long sA, long long sB,
             long long sCinner, long long sCouter, int cmod,
             long long sAdd, float scale)
{
    const int z = blockIdx.z;
    const f16* Az = A + (long long)z * sA;
    const f16* Bz = BT + (long long)z * sB;
    const long long coff = (long long)(z / cmod) * sCouter
                         + (long long)(z % cmod) * sCinner;
    const float* addz = addC ? (addC + (long long)z * sAdd) : nullptr;

    const int wave = threadIdx.x >> 5;
    const int lane = threadIdx.x & 31;
    const int l16  = lane & 15;
    const int lhi  = lane >> 4;          // 0 or 1

    const int tileM = (blockIdx.y * 8 + wave) * 16;
    const int tileN = blockIdx.x * 64;
    if (tileM >= M) return;              // wave-uniform: EXEC stays all-ones

    v8f acc[4] = {};

    // A fragment gather: lane holds row M = l16; K chunks {kk+sel, kk+16+sel}
    const f16* arow = Az + (long long)(tileM + l16) * lda;

    for (int kk = 0; kk < K; kk += 32) {
        HFrag a;
        const f16* ap = arow + kk + lhi * 8;
        a.h[0] = *(const v8h*)(ap);
        a.h[1] = *(const v8h*)(ap + 16);
#pragma unroll
        for (int t = 0; t < 4; ++t) {
            // B fragment: lane = column N = tileN+16t+l16, K chunks {kk+8*lhi, kk+16+8*lhi} (8 each)
            const f16* bp = Bz + (long long)(tileN + t * 16 + l16) * ldb
                          + kk + lhi * 8;
            HFrag bf;
            bf.h[0] = *(const v8h*)(bp);
            bf.h[1] = *(const v8h*)(bp + 16);
            acc[t] = __builtin_amdgcn_wmma_f32_16x16x32_f16(
                false, a.v, false, bf.v, (short)0, acc[t], false, false);
        }
    }

    // C/D layout: VGPR g -> M = g + 8*lhi, N = l16 (within 16x16 tile)
#pragma unroll
    for (int t = 0; t < 4; ++t) {
        const int col = tileN + t * 16 + l16;
#pragma unroll
        for (int g = 0; g < 8; ++g) {
            const int row = tileM + g + lhi * 8;
            float val = acc[t][g] * scale;
            if (addz) val += addz[(long long)row * N + col];
            if (bias) val += bias[col];
            const long long ci = coff + (long long)row * ldc + col;
            if (Cf) Cf[ci] = val;
            else    Ch[ci] = (f16)val;
        }
    }
}

// --------------------------- small helper kernels ---------------------------

__global__ void cvt_f32_f16(const float* __restrict__ src, f16* __restrict__ dst,
                            long long n)
{
    long long i = (long long)blockIdx.x * blockDim.x + threadIdx.x;
    if (i < n) dst[i] = (f16)src[i];
}

// dst[z][c][r] = (f16) src[z][r][c]   (src: [R][C] f32, dst: [C][R] f16)
__global__ void transpose_f32_f16(const float* __restrict__ src,
                                  f16* __restrict__ dst, int R, int C)
{
    const long long bs = (long long)blockIdx.z * R * C;
    const int idx = blockIdx.x * blockDim.x + threadIdx.x;
    if (idx >= R * C) return;
    const int c = idx / R;
    const int r = idx - c * R;
    dst[bs + (long long)c * R + r] = (f16)src[bs + (long long)r * C + c];
}

// beta[m][h] = x[m,:] . wbeta[:,h] + bbeta[h]
__global__ void beta_kernel(const float* __restrict__ x,
                            const float* __restrict__ wbeta,
                            const float* __restrict__ bbeta,
                            float* __restrict__ beta)
{
    const int idx = blockIdx.x * blockDim.x + threadIdx.x;
    if (idx >= M_ * H_) return;
    const int m = idx >> 3, h = idx & 7;
    const float* xr = x + (long long)m * DIM_;
    float s = bbeta[h];
    for (int k = 0; k < DIM_; ++k) s = fmaf(xr[k], wbeta[k * H_ + h], s);
    beta[idx] = s;
}

__device__ __forceinline__ float xcat_elem(const f16* p, int j)
{
    float v = (j < D_) ? (float)p[j] : -(float)p[j - D_];
    return fmaxf(v, 0.0f);
}

// dpfp features + L1 normalize for q and k. One 128-thread block per (bh, n).
// Writes qf[bh][n][128], kf[bh][n][128], and kfT[bh][128][4096] (for W update).
__global__ void __launch_bounds__(128)
features_kernel(const f16* __restrict__ qkv,
                f16* __restrict__ qf, f16* __restrict__ kf,
                f16* __restrict__ kfT)
{
    __shared__ float red[128];
    const int n  = blockIdx.x;            // 0..4095
    const int bh = blockIdx.y;            // 0..31
    const int b = bh >> 3, h = bh & 7;
    const long long m = (long long)b * N_ + n;
    const f16* row = qkv + m * QKV_;
    const int t = threadIdx.x;
    const int jm1 = (t + S_ - 1) & (S_ - 1);

    // ---- q features ----
    const f16* qp = row + h * D_;
    float fq = xcat_elem(qp, t) * xcat_elem(qp, jm1);
    red[t] = fq;
    __syncthreads();
    for (int s = 64; s > 0; s >>= 1) {
        if (t < s) red[t] += red[t + s];
        __syncthreads();
    }
    const float qs = red[0];
    __syncthreads();

    // ---- k features ----
    const f16* kp = row + DIM_ + h * D_;
    float fk = xcat_elem(kp, t) * xcat_elem(kp, jm1);
    red[t] = fk;
    __syncthreads();
    for (int s = 64; s > 0; s >>= 1) {
        if (t < s) red[t] += red[t + s];
        __syncthreads();
    }
    const float ks = red[0];

    const long long fo = ((long long)bh * N_ + n) * S_ + t;
    qf[fo] = (f16)(fq / qs);
    kf[fo] = (f16)(fk / ks);
    kfT[((long long)bh * S_ + t) * N_ + n] = (f16)(fk / ks);
}

// dvT[bh][d][n] = beta[m][h] * (v[m][h*64+d] - vo[bh][n][d])
__global__ void dv_kernel(const f16* __restrict__ qkv,
                          const float* __restrict__ vo,
                          const float* __restrict__ beta,
                          f16* __restrict__ dvT)
{
    const long long idx = (long long)blockIdx.x * blockDim.x + threadIdx.x;
    if (idx >= (long long)32 * N_ * D_) return;
    const int d = (int)(idx & (D_ - 1));
    const long long r = idx >> 6;          // bh*4096 + n
    const int n  = (int)(r & (N_ - 1));
    const int bh = (int)(r >> 12);
    const int b = bh >> 3, h = bh & 7;
    const long long m = (long long)b * N_ + n;
    const float v = (float)qkv[m * QKV_ + 2 * DIM_ + h * D_ + d];
    const float dv = beta[m * H_ + h] * (v - vo[idx]);
    dvT[((long long)bh * D_ + d) * N_ + n] = (f16)dv;
}

// ---------------------------------------------------------------------------

extern "C" void kernel_launch(void* const* d_in, const int* in_sizes, int n_in,
                              void* d_out, int out_size, void* d_ws, size_t ws_size,
                              hipStream_t stream)
{
    const float* x     = (const float*)d_in[0];
    const float* W     = (const float*)d_in[1];   // [4,8,128,64]
    const float* wqkv  = (const float*)d_in[2];   // [512,1536]
    const float* wbeta = (const float*)d_in[3];   // [512,8]
    const float* bbeta = (const float*)d_in[4];   // [8]
    const float* wout  = (const float*)d_in[5];   // [512,512]
    const float* bout  = (const float*)d_in[6];   // [512]

    float* out  = (float*)d_out;                       // [4,4096,512]
    float* Wnew = out + (long long)M_ * DIM_;          // [4,8,128,64]

    char* ws = (char*)d_ws;
    size_t cur = 0;
    auto alloc = [&](size_t bytes) -> char* {
        char* p = ws + cur;
        cur = (cur + bytes + 255) & ~(size_t)255;
        return p;
    };

    f16*   xh     = (f16*)  alloc((size_t)M_ * DIM_ * 2);        // x in f16
    f16*   wqkvT  = (f16*)  alloc((size_t)QKV_ * DIM_ * 2);      // [1536][512]
    f16*   woutT  = (f16*)  alloc((size_t)DIM_ * DIM_ * 2);      // [512][512]
    f16*   WT     = (f16*)  alloc((size_t)32 * D_ * S_ * 2);     // [bh][64][128]
    f16*   qkvb   = (f16*)  alloc((size_t)M_ * QKV_ * 2);        // [16384][1536]
    float* betab  = (float*)alloc((size_t)M_ * H_ * 4);          // [16384][8]
    f16*   qf     = (f16*)  alloc((size_t)32 * N_ * S_ * 2);     // [bh][4096][128]
    f16*   kf     = (f16*)  alloc((size_t)32 * N_ * S_ * 2);
    f16*   kfT    = (f16*)  alloc((size_t)32 * S_ * N_ * 2);     // [bh][128][4096]
    float* vo     = (float*)alloc((size_t)32 * N_ * D_ * 4);     // [bh][4096][64]
    f16*   dvT    = (f16*)  alloc((size_t)32 * D_ * N_ * 2);     // [bh][64][4096]
    f16*   WnewT  = (f16*)  alloc((size_t)32 * D_ * S_ * 2);     // [bh][64][128]
    f16*   A2     = (f16*)  alloc((size_t)M_ * DIM_ * 2);        // [16384][512]

    const long long BIG = 1LL << 40;  (void)BIG; (void)ws_size; (void)in_sizes;
    (void)n_in; (void)out_size;

    // 1) f16 conversions / transposes of the small weights (L2 resident).
    cvt_f32_f16<<<dim3((M_ * DIM_ + 255) / 256), 256, 0, stream>>>(
        x, xh, (long long)M_ * DIM_);
    transpose_f32_f16<<<dim3((DIM_ * QKV_ + 255) / 256, 1, 1), 256, 0, stream>>>(
        wqkv, wqkvT, DIM_, QKV_);
    transpose_f32_f16<<<dim3((DIM_ * DIM_ + 255) / 256, 1, 1), 256, 0, stream>>>(
        wout, woutT, DIM_, DIM_);
    transpose_f32_f16<<<dim3((S_ * D_ + 255) / 256, 1, 32), 256, 0, stream>>>(
        W, WT, S_, D_);

    // 2) qkv = x @ wqkv   (M=16384, N=1536, K=512) -> f16
    wmma_gemm_bt<<<dim3(QKV_ / 64, M_ / 128, 1), 256, 0, stream>>>(
        xh, wqkvT, nullptr, qkvb, nullptr, nullptr,
        M_, QKV_, DIM_, DIM_, DIM_, QKV_,
        0LL, 0LL, 0LL, 0LL, 1, 0LL, 1.0f);

    // 3) beta = x @ wbeta + bbeta
    beta_kernel<<<dim3((M_ * H_ + 255) / 256), 256, 0, stream>>>(
        x, wbeta, bbeta, betab);

    // 4) dpfp features + normalize (qf, kf, kfT)
    features_kernel<<<dim3(N_, 32, 1), 128, 0, stream>>>(qkvb, qf, kf, kfT);

    // 5) vo = kf @ W   (per head: M=4096, N=64, K=128)
    wmma_gemm_bt<<<dim3(1, N_ / 128, 32), 256, 0, stream>>>(
        kf, WT, vo, nullptr, nullptr, nullptr,
        N_, D_, S_, S_, S_, D_,
        (long long)N_ * S_, (long long)D_ * S_,
        0LL, (long long)N_ * D_, 1, 0LL, 1.0f);

    // 6) dvT = beta * (v - vo), transposed to [bh][64][4096]
    dv_kernel<<<dim3((32 * N_ * D_ + 255) / 256), 256, 0, stream>>>(
        qkvb, vo, betab, dvT);

    // 7) W_new = W + kf^T @ dv   (per head: M=128, N=64, K=4096) -> d_out tail
    wmma_gemm_bt<<<dim3(1, 1, 32), 256, 0, stream>>>(
        kfT, dvT, Wnew, nullptr, W, nullptr,
        S_, D_, N_, N_, N_, D_,
        (long long)S_ * N_, (long long)D_ * N_,
        0LL, (long long)S_ * D_, 1, (long long)S_ * D_, 1.0f);

    // 8) W_new^T in f16 for the output GEMM
    transpose_f32_f16<<<dim3((S_ * D_ + 255) / 256, 1, 32), 256, 0, stream>>>(
        Wnew, WnewT, S_, D_);

    // 9) outh = (qf @ W_new) * SCALE  scattered into A2[b][n][h*64+d] (f16)
    wmma_gemm_bt<<<dim3(1, N_ / 128, 32), 256, 0, stream>>>(
        qf, WnewT, nullptr, A2, nullptr, nullptr,
        N_, D_, S_, S_, S_, DIM_,
        (long long)N_ * S_, (long long)D_ * S_,
        (long long)D_, (long long)N_ * DIM_, 8, 0LL, 0.125f);

    // 10) out = A2 @ wout + bout   (M=16384, N=512, K=512) -> d_out head
    wmma_gemm_bt<<<dim3(DIM_ / 64, M_ / 128, 1), 256, 0, stream>>>(
        A2, woutT, out, nullptr, nullptr, bout,
        M_, DIM_, DIM_, DIM_, DIM_, DIM_,
        0LL, 0LL, 0LL, 0LL, 1, 0LL, 1.0f);
}
